// PT_28140625723964
// MI455X (gfx1250) — matrix operand order repaired and checked by reading, not compiled, not run on hardware
//
#include <hip/hip_runtime.h>

typedef _Float16 half_t;
typedef __attribute__((ext_vector_type(8)))  _Float16 v8h;
typedef __attribute__((ext_vector_type(16))) _Float16 v16h;
typedef __attribute__((ext_vector_type(8)))  float    v8f;

#define BATCH 8192
#define LHIST 100
#define NTOP  20
#define DVEC  100
#define FEATK 64   // 60 padded to 64 (two WMMA k-steps of 32)
#define FCN   32   // 20 padded to 32 (two 16-wide N tiles)

__device__ __forceinline__ float wave_sum(float v) {
  #pragma unroll
  for (int m = 16; m; m >>= 1) v += __shfl_xor(v, m, 32);
  return v;
}

// Computes the prospect-theory gain for one (batch,item) row; all 32 lanes of a
// wave cooperate. lane<20 also returns its lda element in *lda_out.
__device__ __forceinline__ float row_gain(
    int lane,
    const float* __restrict__ vec_row, const float* __restrict__ lda_row,
    const float* __restrict__ part_row, const float* __restrict__ info_row,
    const float* __restrict__ auth_row, const float* __restrict__ inter_row,
    const float* __restrict__ s_uvec, const float* __restrict__ s_ulda,
    const float* __restrict__ s_upart, const float* __restrict__ s_norm,
    const float* __restrict__ s_param, float* lda_out) {
  const float EPS = 1e-8f;
  float vdot = 0.f, vsq = 0.f, ldot = 0.f, lsq = 0.f, pdot = 0.f, psq = 0.f;
  float info = 0.f, auth = 0.f, inter = 0.f;
  if (lane < 25) {  // 100-dim vector: float4 per lane, coalesced b128 loads
    const float4 v  = *(const float4*)(vec_row + 4 * lane);
    const float4 uv = *(const float4*)(s_uvec  + 4 * lane);
    vdot = v.x * uv.x + v.y * uv.y + v.z * uv.z + v.w * uv.w;
    vsq  = v.x * v.x  + v.y * v.y  + v.z * v.z  + v.w * v.w;
  }
  if (lane < NTOP) {
    const float lv = lda_row[lane];
    *lda_out = lv;
    ldot = lv * s_ulda[lane];
    lsq  = lv * lv;
    const float pv = part_row[lane];
    pdot = pv * s_upart[lane];
    psq  = pv * pv;
  }
  if (lane < 3) {
    info = info_row[lane] * s_param[11 + lane];
    auth = auth_row[lane] * s_param[16 + lane];
  }
  if (lane < 2) inter = inter_row[lane] * s_param[14 + lane];
  vdot = wave_sum(vdot); vsq = wave_sum(vsq);
  ldot = wave_sum(ldot); lsq = wave_sum(lsq);
  pdot = wave_sum(pdot); psq = wave_sum(psq);
  info = wave_sum(info); auth = wave_sum(auth); inter = wave_sum(inter);
  const float lda_gain = ldot / fmaxf(__builtin_sqrtf(lsq) * s_norm[1], EPS);
  const float vec_gain = vdot / fmaxf(__builtin_sqrtf(vsq) * s_norm[0], EPS);
  const float part_sim = pdot / fmaxf(__builtin_sqrtf(psq) * s_norm[2], EPS);
  const float x = lda_gain * s_param[1] + vec_gain * s_param[2]
                + info * s_param[3] + part_sim * s_param[4]
                + inter * s_param[5] + auth * s_param[6] - s_param[7];
  const bool pos = (x > 0.f);
  const float e = pos ? s_param[9] : s_param[10];
  const float v = __powf(fabsf(x), e);
  return pos ? v : -s_param[8] * v;
}

__global__ __launch_bounds__(256) void main_kernel(
    const int* __restrict__ user,
    const float* __restrict__ hist_lda, const float* __restrict__ hist_vector,
    const float* __restrict__ hist_info, const float* __restrict__ hist_authority,
    const float* __restrict__ hist_participants, const float* __restrict__ hist_interact,
    const float* __restrict__ timeDelta,
    const float* __restrict__ item_lda, const float* __restrict__ item_vector,
    const float* __restrict__ item_info, const float* __restrict__ item_authority,
    const float* __restrict__ item_participants, const float* __restrict__ item_interact,
    const float* __restrict__ know_lda_pref, const float* __restrict__ know_vector_pref,
    const float* __restrict__ lda_gain_ref, const float* __restrict__ com_participant_pref,
    const float* __restrict__ td_g, const float* __restrict__ td_u,
    const float* __restrict__ info_part_g, const float* __restrict__ info_part_u,
    const float* __restrict__ topic_g, const float* __restrict__ topic_u,
    const float* __restrict__ content_g, const float* __restrict__ content_u,
    const float* __restrict__ info_w_u,
    const float* __restrict__ inter_apart_g, const float* __restrict__ inter_apart_u,
    const float* __restrict__ auth_apart_g, const float* __restrict__ auth_apart_u,
    const float* __restrict__ part_w_g, const float* __restrict__ part_w_u,
    const float* __restrict__ inter_w_g, const float* __restrict__ inter_w_u,
    const float* __restrict__ auth_w_g, const float* __restrict__ auth_w_u,
    const float* __restrict__ xref_g, const float* __restrict__ xref_u,
    const float* __restrict__ xlam_g, const float* __restrict__ xlam_u,
    const float* __restrict__ xalpha_g, const float* __restrict__ xalpha_u,
    const float* __restrict__ xbeta_g, const float* __restrict__ xbeta_u,
    half_t* __restrict__ feat_out) {
  __shared__ __align__(16) float s_uvec[DVEC];
  __shared__ float s_ulda[NTOP];
  __shared__ float s_upart[NTOP];
  __shared__ float s_norm[3];
  __shared__ float s_param[19];
  __shared__ float s_hist_topic[NTOP];

  const int b   = blockIdx.x;
  const int tid = threadIdx.x;
  const int lane = tid & 31;
  const int wid  = tid >> 5;
  const int u = user[b];  // uniform per block -> scalar load

  // ---- phase 0: cache user preference vectors + fold per-user scalars ----
  if (tid < DVEC) {
    s_uvec[tid] = know_vector_pref[(size_t)u * DVEC + tid];
  } else if (tid < DVEC + NTOP) {
    s_ulda[tid - DVEC] = know_lda_pref[(size_t)u * NTOP + (tid - DVEC)];
  } else if (tid < DVEC + 2 * NTOP) {
    s_upart[tid - DVEC - NTOP] = com_participant_pref[(size_t)u * NTOP + (tid - DVEC - NTOP)];
  } else if (tid < DVEC + 3 * NTOP) {
    s_hist_topic[tid - DVEC - 2 * NTOP] = 0.f;
  } else if (tid == 192) {
    s_param[0]  = td_g[0]      + td_u[u];
    s_param[1]  = topic_g[0]   + topic_u[u];
    s_param[2]  = content_g[0] + content_u[u];
    s_param[3]  = info_w_u[0]  + info_w_u[u];   // faithful quirk: row 0 + user row
    s_param[4]  = part_w_g[0]  + part_w_u[u];
    s_param[5]  = inter_w_g[0] + inter_w_u[u];
    s_param[6]  = auth_w_g[0]  + auth_w_u[u];
    s_param[7]  = xref_g[0]    + xref_u[u];
    s_param[8]  = xlam_g[0]    + xlam_u[u];
    s_param[9]  = xalpha_g[0]  + xalpha_u[u];
    s_param[10] = xbeta_g[0]   + xbeta_u[u];
    #pragma unroll
    for (int i = 0; i < 3; ++i) s_param[11 + i] = info_part_g[i]  + info_part_u[(size_t)u * 3 + i];
    #pragma unroll
    for (int i = 0; i < 2; ++i) s_param[14 + i] = inter_apart_g[i] + inter_apart_u[(size_t)u * 2 + i];
    #pragma unroll
    for (int i = 0; i < 3; ++i) s_param[16 + i] = auth_apart_g[i]  + auth_apart_u[(size_t)u * 3 + i];
  }
  __syncthreads();
  if (tid < 3) {  // norms of the cached user vectors
    float s = 0.f;
    if (tid == 0) { for (int i = 0; i < DVEC; ++i) s += s_uvec[i] * s_uvec[i]; }
    else if (tid == 1) { for (int i = 0; i < NTOP; ++i) s += s_ulda[i] * s_ulda[i]; }
    else { for (int i = 0; i < NTOP; ++i) s += s_upart[i] * s_upart[i]; }
    s_norm[tid] = __builtin_sqrtf(s);
  }
  __syncthreads();

  // ---- phase 1: stream history, one item per wave-iteration ----
  float acc = 0.f;
  for (int l = wid; l < LHIST; l += 8) {
    const size_t r = (size_t)b * LHIST + l;
    float myLda = 0.f;
    const float g = row_gain(lane,
        hist_vector + r * DVEC, hist_lda + r * NTOP, hist_participants + r * NTOP,
        hist_info + r * 3, hist_authority + r * 3, hist_interact + r * 2,
        s_uvec, s_ulda, s_upart, s_norm, s_param, &myLda);
    const float w = __expf(-timeDelta[r] * s_param[0]);
    if (lane < NTOP) acc += myLda * (g * w);
  }
  if (lane < NTOP) atomicAdd(&s_hist_topic[lane], acc);
  __syncthreads();

  // ---- phase 2: current item + feature emit (wave 0) ----
  if (wid == 0) {
    float myItemLda = 0.f;
    const float cg = row_gain(lane,
        item_vector + (size_t)b * DVEC, item_lda + (size_t)b * NTOP,
        item_participants + (size_t)b * NTOP, item_info + (size_t)b * 3,
        item_authority + (size_t)b * 3, item_interact + (size_t)b * 2,
        s_uvec, s_ulda, s_upart, s_norm, s_param, &myItemLda);
    half_t* f = feat_out + (size_t)b * FEATK;
    if (lane < NTOP) {
      const float curr = cg * myItemLda;
      const float diff = lda_gain_ref[(size_t)u * NTOP + lane] - s_hist_topic[lane];
      f[lane]             = (half_t)diff;
      f[NTOP + lane]      = (half_t)(diff * curr);
      f[2 * NTOP + lane]  = (half_t)curr;
    }
    if (lane < FEATK - 3 * NTOP) f[3 * NTOP + lane] = (half_t)0.f;  // K padding
  }
}

// Pack fc1_w [20,60] (row-major out,in) into the per-lane WMMA B-fragment
// layout so fc_wmma_kernel can fetch each lane's v16h with contiguous b128
// loads. Linear index: ((nt*2 + s)*32 + lane)*16 + j, where
//   K = (j&7) + 8*(lane>>4) + 16*(j>>3) + 32*s,  N = nt*16 + (lane&15)  (zeros in padding).
__global__ void pack_w_kernel(const float* __restrict__ fc1_w, half_t* __restrict__ wpack) {
  const int idx = blockIdx.x * blockDim.x + threadIdx.x;
  if (idx < 2 * 2 * 32 * 16) {
    const int j    = idx & 15;
    const int lane = (idx >> 4) & 31;
    const int s    = (idx >> 9) & 1;
    const int nt   = idx >> 10;
    const int k = (j & 7) + 8 * (lane >> 4) + 16 * (j >> 3) + 32 * s;
    const int n = nt * 16 + (lane & 15);
    const float v = (k < 60 && n < NTOP) ? fc1_w[n * 60 + k] : 0.f;
    wpack[idx] = (half_t)v;
  }
}

// feat[8192,64]f16 x W[64,32]f16 -> h[8192,32]f32 via v_wmma_f32_16x16x32_f16,
// fused fc1_b + fc2 epilogue -> out[8192]. One wave per 16-row M tile.
__global__ __launch_bounds__(128) void fc_wmma_kernel(
    const half_t* __restrict__ feat, const half_t* __restrict__ wpack,
    const float* __restrict__ fc1_b, const float* __restrict__ fc2_w,
    const float* __restrict__ fc2_b, float* __restrict__ out) {
  const int wave = (blockIdx.x * blockDim.x + threadIdx.x) >> 5;  // = M tile index
  const int lane = threadIdx.x & 31;
  const int m    = lane & 15;
  const int hh   = lane >> 4;        // lane half (ISA 16-bit A/B layouts)
  const int row0 = wave * 16;

  const half_t* bbase = wpack + (size_t)lane * 16;  // lane's swizzled B fragments

  v8f c0 = {}, c1 = {};
  #pragma unroll
  for (int s = 0; s < 2; ++s) {      // K steps: 0..31, 32..63
    // A: 16x32 f16, lane m = row; ISA layout = two contiguous 8-half chunks
    //   chunk0: K = 8*hh + 32*s + [0..7],  chunk1: K = 16 + 8*hh + 32*s + [0..7]
    const half_t* arow = feat + (size_t)(row0 + m) * FEATK + 8 * hh + 32 * s;
    const v8h alo = *(const v8h*)(arow);
    const v8h ahi = *(const v8h*)(arow + 16);
    const v16h a = __builtin_shufflevector(alo, ahi,
        0, 1, 2, 3, 4, 5, 6, 7, 8, 9, 10, 11, 12, 13, 14, 15);
    // B: pre-swizzled fragments, contiguous 32B per lane -> b128 pairs
    const v16h b0 = *(const v16h*)(bbase + (size_t)s * 512);          // N tile 0
    const v16h b1 = *(const v16h*)(bbase + 1024 + (size_t)s * 512);   // N tile 1
    c0 = __builtin_amdgcn_wmma_f32_16x16x32_f16(false, a, false, b0, (short)0, c0, false, false);
    c1 = __builtin_amdgcn_wmma_f32_16x16x32_f16(false, a, false, b1, (short)0, c1, false, false);
  }

  // epilogue: out[row] = sum_n (h[row][n] + fc1_b[n]) * fc2_w[n] + fc2_b
  const float w0 = fc2_w[m], bias0 = fc1_b[m];          // n = m < 20
  const int n1 = 16 + m;
  const float w1    = (n1 < NTOP) ? fc2_w[n1] : 0.f;    // padded cols killed by w1=0
  const float bias1 = (n1 < NTOP) ? fc1_b[n1] : 0.f;
  float partial[8];
  #pragma unroll
  for (int j = 0; j < 8; ++j)
    partial[j] = (c0[j] + bias0) * w0 + (c1[j] + bias1) * w1;
  #pragma unroll
  for (int mask = 1; mask < 16; mask <<= 1)
    #pragma unroll
    for (int j = 0; j < 8; ++j)
      partial[j] += __shfl_xor(partial[j], mask, 32);
  if (m == 0) {
    const float b2 = fc2_b[0];
    #pragma unroll
    for (int j = 0; j < 8; ++j)
      out[row0 + j + 8 * hh] = partial[j] + b2;   // lanes 0/16 cover rows 0-7 / 8-15
  }
}

extern "C" void kernel_launch(void* const* d_in, const int* in_sizes, int n_in,
                              void* d_out, int out_size, void* d_ws, size_t ws_size,
                              hipStream_t stream) {
  const int*   user         = (const int*)  d_in[0];
  const float* hist_lda     = (const float*)d_in[1];
  const float* hist_vector  = (const float*)d_in[2];
  const float* hist_info    = (const float*)d_in[3];
  const float* hist_auth    = (const float*)d_in[4];
  const float* hist_part    = (const float*)d_in[5];
  const float* hist_inter   = (const float*)d_in[6];
  const float* timeDelta    = (const float*)d_in[7];
  const float* item_lda     = (const float*)d_in[8];
  const float* item_vector  = (const float*)d_in[9];
  const float* item_info    = (const float*)d_in[10];
  const float* item_auth    = (const float*)d_in[11];
  const float* item_part    = (const float*)d_in[12];
  const float* item_inter   = (const float*)d_in[13];
  const float* know_lda     = (const float*)d_in[14];
  const float* know_vec     = (const float*)d_in[15];
  const float* lda_gain_ref = (const float*)d_in[16];
  const float* com_part     = (const float*)d_in[17];
  const float* td_g         = (const float*)d_in[18];
  const float* td_u         = (const float*)d_in[19];
  const float* info_part_g  = (const float*)d_in[20];
  const float* info_part_u  = (const float*)d_in[21];
  const float* topic_g      = (const float*)d_in[22];
  const float* topic_u      = (const float*)d_in[23];
  const float* content_g    = (const float*)d_in[24];
  const float* content_u    = (const float*)d_in[25];
  const float* info_w_u     = (const float*)d_in[26];
  const float* inter_ap_g   = (const float*)d_in[27];
  const float* inter_ap_u   = (const float*)d_in[28];
  const float* auth_ap_g    = (const float*)d_in[29];
  const float* auth_ap_u    = (const float*)d_in[30];
  const float* part_w_g     = (const float*)d_in[31];
  const float* part_w_u     = (const float*)d_in[32];
  const float* inter_w_g    = (const float*)d_in[33];
  const float* inter_w_u    = (const float*)d_in[34];
  const float* auth_w_g     = (const float*)d_in[35];
  const float* auth_w_u     = (const float*)d_in[36];
  const float* xref_g       = (const float*)d_in[37];
  const float* xref_u       = (const float*)d_in[38];
  const float* xlam_g       = (const float*)d_in[39];
  const float* xlam_u       = (const float*)d_in[40];
  const float* xalpha_g     = (const float*)d_in[41];
  const float* xalpha_u     = (const float*)d_in[42];
  const float* xbeta_g      = (const float*)d_in[43];
  const float* xbeta_u      = (const float*)d_in[44];
  const float* fc1_w        = (const float*)d_in[45];
  const float* fc1_b        = (const float*)d_in[46];
  const float* fc2_w        = (const float*)d_in[47];
  const float* fc2_b        = (const float*)d_in[48];

  half_t* feat  = (half_t*)d_ws;                                   // 8192*64*2 = 1 MB
  half_t* wpack = (half_t*)((char*)d_ws + (size_t)BATCH * FEATK * sizeof(half_t)); // 4 KB

  pack_w_kernel<<<(2 * 2 * 32 * 16 + 255) / 256, 256, 0, stream>>>(fc1_w, wpack);

  main_kernel<<<BATCH, 256, 0, stream>>>(
      user, hist_lda, hist_vector, hist_info, hist_auth, hist_part, hist_inter,
      timeDelta, item_lda, item_vector, item_info, item_auth, item_part, item_inter,
      know_lda, know_vec, lda_gain_ref, com_part,
      td_g, td_u, info_part_g, info_part_u, topic_g, topic_u, content_g, content_u,
      info_w_u, inter_ap_g, inter_ap_u, auth_ap_g, auth_ap_u,
      part_w_g, part_w_u, inter_w_g, inter_w_u, auth_w_g, auth_w_u,
      xref_g, xref_u, xlam_g, xlam_u, xalpha_g, xalpha_u, xbeta_g, xbeta_u,
      feat);

  // 512 M-tiles, one wave each -> 128 blocks of 4 waves
  fc_wmma_kernel<<<(BATCH / 16) / 4, 128, 0, stream>>>(
      feat, wpack, fc1_b, fc2_w, fc2_b, (float*)d_out);
}
